// TransformerEncoder_70239895158960
// MI455X (gfx1250) — compile-verified, matrix operand order checked by reading, not run on hardware
//
#include <hip/hip_runtime.h>
#include <math.h>

// Problem constants (match reference)
#define BB 8
#define SS 1024
#define DD 768
#define HH 12
#define FF 3072
#define LL 6
#define DKH 64   // head dim

typedef __attribute__((ext_vector_type(16))) __bf16 v16bf;
typedef __attribute__((ext_vector_type(8)))  __bf16 v8bf;
typedef __attribute__((ext_vector_type(4)))  __bf16 v4bf;
typedef __attribute__((ext_vector_type(8)))  float  v8f;

__device__ __forceinline__ __bf16 f2bf(float f) { return (__bf16)f; }

// ---- CDNA5 async global->LDS copy (ASYNCcnt-tracked), 16 bytes per lane ----
__device__ __forceinline__ void async_copy16(const __bf16* lds_dst,
                                             const __bf16* gsrc) {
  unsigned l = (unsigned)(unsigned long long)lds_dst;   // low 32b = LDS offset
  unsigned long long g = (unsigned long long)gsrc;
  asm volatile("global_load_async_to_lds_b128 %0, %1, off"
               :: "v"(l), "v"(g) : "memory");
}
__device__ __forceinline__ void wait_async_le6() {
  asm volatile("s_wait_asynccnt 0x6" ::: "memory");
}
__device__ __forceinline__ void wait_async_0() {
  asm volatile("s_wait_asynccnt 0x0" ::: "memory");
}

// Load a 16x32 bf16 fragment (A-layout per CDNA5 ISA 7.12.2) from LDS.
__device__ __forceinline__ v16bf load_frag(const __bf16* __restrict__ base, int ld) {
  int lane = threadIdx.x & 31;
  const __bf16* p = base + (lane & 15) * ld + ((lane >> 4) << 3);
  v8bf lo = *(const v8bf*)p;          // ds_load_b128
  v8bf hi = *(const v8bf*)(p + 16);
  v16bf r;
#pragma unroll
  for (int i = 0; i < 8; ++i) { r[i] = lo[i]; r[i + 8] = hi[i]; }
  return r;
}

__device__ __forceinline__ v8f wmma_bf16(v16bf a, v16bf b, v8f c) {
  return __builtin_amdgcn_wmma_f32_16x16x32_bf16(false, a, false, b, (short)0, c,
                                                 false, false);
}

// ---------------------------------------------------------------------------
// fp32 -> bf16 bulk convert (float4 -> 4x bf16 per thread)
// ---------------------------------------------------------------------------
__global__ __launch_bounds__(256)
void f32_to_bf16(const float* __restrict__ in, __bf16* __restrict__ out, int n4) {
  int i = blockIdx.x * 256 + threadIdx.x;
  if (i < n4) {
    float4 v = ((const float4*)in)[i];
    v4bf b = { f2bf(v.x), f2bf(v.y), f2bf(v.z), f2bf(v.w) };
    ((v4bf*)out)[i] = b;
  }
}

// ---------------------------------------------------------------------------
// LayerNorm over the SEQ axis (axis=1), unbiased std, eps added to std.
// Reads fp32 residual stream, writes bf16. grid (DD/32, BB), block (32, 8)
// ---------------------------------------------------------------------------
__global__ __launch_bounds__(256)
void layernorm_seq(const float* __restrict__ x, const float* __restrict__ alpha,
                   const float* __restrict__ beta, __bf16* __restrict__ out) {
  __shared__ float rs[8][32];
  __shared__ float rq[8][32];
  __shared__ float sc[32];
  __shared__ float sh[32];
  int tx = threadIdx.x, ty = threadIdx.y;
  int b = blockIdx.y;
  int d = blockIdx.x * 32 + tx;
  float s = 0.f, q = 0.f;
  for (int si = ty; si < SS; si += 8) {
    float v = x[((size_t)b * SS + si) * DD + d];
    s += v; q += v * v;
  }
  rs[ty][tx] = s; rq[ty][tx] = q;
  __syncthreads();
  if (ty == 0) {
    float S1 = 0.f, S2 = 0.f;
#pragma unroll
    for (int i = 0; i < 8; ++i) { S1 += rs[i][tx]; S2 += rq[i][tx]; }
    float mean = S1 / (float)SS;
    float var  = (S2 - (float)SS * mean * mean) / (float)(SS - 1);
    var = fmaxf(var, 0.f);
    float scale = alpha[d] / (sqrtf(var) + 1e-6f);
    sc[tx] = scale;
    sh[tx] = beta[d] - mean * scale;
  }
  __syncthreads();
  float scale = sc[tx], shift = sh[tx];
  for (int si = ty; si < SS; si += 8) {
    size_t idx = ((size_t)b * SS + si) * DD + d;
    out[idx] = f2bf(x[idx] * scale + shift);
  }
}

// ---------------------------------------------------------------------------
// C(MxN) = A(MxK) @ W(NxK)^T, bf16 in / bf16 WMMA / fp32 accum.
// flags: bit0 = GELU, bit1 = bf16 output (else fp32 + optional residual)
// Block tile 256x128, 8 waves (4x2), wave tile 64x64 (16 WMMA / 16 ds_b128
// per k-step). Double-buffered LDS via GLOBAL_LOAD_ASYNC_TO_LDS_B128.
// grid (N/128, M/256), block 256.
// ---------------------------------------------------------------------------
__global__ __launch_bounds__(256)
void gemm_nt_bf16(const __bf16* __restrict__ A, const __bf16* __restrict__ W,
                  void* __restrict__ Cout, const float* __restrict__ bias,
                  const float* __restrict__ residual,
                  int M, int N, int K, int flags) {
  __shared__ __align__(16) __bf16 As[2][256][40];   // 80B rows (16B multiple)
  __shared__ __align__(16) __bf16 Bs[2][128][40];
  int tid = threadIdx.x;
  int lane = tid & 31, wid = tid >> 5;
  int wm = wid & 3, wn = wid >> 2;                  // 4 x 2 wave grid
  int bm = blockIdx.y * 256, bn = blockIdx.x * 128;
  int r0 = tid >> 2;            // 0..63
  int c0 = (tid & 3) << 3;      // bf16 cols 0,8,16,24 (16B chunks)

  const int nk = K >> 5;

  // async-fill one buffer: A = 4 chunks (256 rows), B = 2 chunks (128 rows)
  auto issue_tile = [&](int buf, int kc) {
    const __bf16* ga = A + (size_t)(bm + r0) * K + kc + c0;
    const __bf16* gw = W + (size_t)(bn + r0) * K + kc + c0;
#pragma unroll
    for (int p = 0; p < 4; ++p)
      async_copy16(&As[buf][r0 + p * 64][c0], ga + (size_t)(p * 64) * K);
#pragma unroll
    for (int p = 0; p < 2; ++p)
      async_copy16(&Bs[buf][r0 + p * 64][c0], gw + (size_t)(p * 64) * K);
  };

  issue_tile(0, 0);

  v8f zero = {};
  v8f acc[4][4];
#pragma unroll
  for (int i = 0; i < 4; ++i)
#pragma unroll
    for (int j = 0; j < 4; ++j) acc[i][j] = zero;

  for (int kt = 0; kt < nk; ++kt) {
    int cur = kt & 1;
    if (kt + 1 < nk) {
      issue_tile(cur ^ 1, (kt + 1) << 5);
      wait_async_le6();         // current tile done; next tile's 6 in flight
    } else {
      wait_async_0();
    }
    __syncthreads();
    v16bf a[4];
#pragma unroll
    for (int mi = 0; mi < 4; ++mi)
      a[mi] = load_frag(&As[cur][wm * 64 + mi * 16][0], 40);
#pragma unroll
    for (int nf = 0; nf < 4; ++nf) {
      v16bf bfrag = load_frag(&Bs[cur][wn * 64 + nf * 16][0], 40);
#pragma unroll
      for (int mi = 0; mi < 4; ++mi)
        acc[mi][nf] = wmma_bf16(a[mi], bfrag, acc[mi][nf]);
    }
    __syncthreads();            // protect buffer `cur` before kt+2 refills it
  }

  int lh = lane >> 4, coll = lane & 15;
#pragma unroll
  for (int mi = 0; mi < 4; ++mi)
#pragma unroll
    for (int nf = 0; nf < 4; ++nf)
#pragma unroll
      for (int j = 0; j < 8; ++j) {
        int row = bm + wm * 64 + mi * 16 + j + 8 * lh;
        int col = bn + wn * 64 + nf * 16 + coll;
        float v = acc[mi][nf][j];
        if (bias) v += bias[col];
        if (flags & 1) v = 0.5f * v * (1.0f + erff(v * 0.70710678f));
        size_t idx = (size_t)row * N + col;
        if (flags & 2) {
          ((__bf16*)Cout)[idx] = f2bf(v);
        } else {
          if (residual) v += residual[idx];
          ((float*)Cout)[idx] = v;
        }
      }
}

// ---------------------------------------------------------------------------
// Flash attention on bf16 q/k/v: per block 128 q-rows of one (b,h),
// streaming 64-key tiles (K via async-to-LDS, V transposed manually).
// grid (SS/128, BB*HH), block 256 (8 waves x 16 q-rows)
// ---------------------------------------------------------------------------
__global__ __launch_bounds__(256)
void flash_attn(const __bf16* __restrict__ Q, const __bf16* __restrict__ Kg,
                const __bf16* __restrict__ Vg, __bf16* __restrict__ O) {
  __shared__ __align__(16) __bf16 Kt[64][72];       // 64 keys x 64 dk (padded)
  __shared__ __align__(16) __bf16 Vt[64][72];       // transposed: [dk][key]
  __shared__ __align__(16) __bf16 Pw[8][16][72];    // per-wave P tile
  int tid = threadIdx.x, lane = tid & 31, wid = tid >> 5;
  int qb = blockIdx.x;
  int b = blockIdx.y / HH, h = blockIdx.y % HH;
  const float scale = 0.125f;                        // 1/sqrt(64)
  size_t headoff = (size_t)h * DKH;

  int r = lane & 15, ko = (lane >> 4) << 3;
  int lh = lane >> 4, coll = lane & 15;

  // This wave's 16 Q rows straight into A-fragments (bf16 16B loads)
  const __bf16* qbase = Q + ((size_t)(b * SS) + (size_t)qb * 128) * DD + headoff;
  v16bf qf[2];
#pragma unroll
  for (int s = 0; s < 2; ++s) {
    const __bf16* p = qbase + (size_t)(wid * 16 + r) * DD + s * 32 + ko;
    v8bf lo = *(const v8bf*)p;
    v8bf hi = *(const v8bf*)(p + 16);
    v16bf f;
#pragma unroll
    for (int i = 0; i < 8; ++i) { f[i] = lo[i]; f[i + 8] = hi[i]; }
    qf[s] = f;
  }

  v8f zero = {};
  v8f o[4];
#pragma unroll
  for (int i = 0; i < 4; ++i) o[i] = zero;
  float mrow[8], lrow[8];
#pragma unroll
  for (int j = 0; j < 8; ++j) { mrow[j] = -3.0e38f; lrow[j] = 0.f; }

  for (int kb = 0; kb < SS / 64; ++kb) {
    const __bf16* kbase = Kg + ((size_t)(b * SS) + (size_t)kb * 64) * DD + headoff;
    const __bf16* vbase = Vg + ((size_t)(b * SS) + (size_t)kb * 64) * DD + headoff;
#pragma unroll
    for (int p = 0; p < 2; ++p) {
      int c = tid + p * 256;            // 512 16B-chunks: 64 rows x 8
      int row = c >> 3, col8 = (c & 7) << 3;
      async_copy16(&Kt[row][col8], kbase + (size_t)row * DD + col8);
      v8bf vv = *(const v8bf*)(vbase + (size_t)row * DD + col8);
#pragma unroll
      for (int j = 0; j < 8; ++j) Vt[col8 + j][row] = vv[j];
    }
    wait_async_0();
    __syncthreads();

    // scores: 16 q-rows x 64 keys = 4 n-fragments, K-dim 64 = 2 wmma steps
    v8f sf[4];
#pragma unroll
    for (int nf = 0; nf < 4; ++nf) {
      v8f c = zero;
      c = wmma_bf16(qf[0], load_frag(&Kt[nf * 16][0], 72), c);
      c = wmma_bf16(qf[1], load_frag(&Kt[nf * 16][32], 72), c);
      sf[nf] = c * scale;
    }

    // online softmax (row reductions within 16-lane halves)
#pragma unroll
    for (int j = 0; j < 8; ++j) {
      float mx = fmaxf(fmaxf(sf[0][j], sf[1][j]), fmaxf(sf[2][j], sf[3][j]));
#pragma unroll
      for (int d = 1; d < 16; d <<= 1) mx = fmaxf(mx, __shfl_xor(mx, d, 32));
      float mnew = fmaxf(mrow[j], mx);
      float psum = 0.f;
#pragma unroll
      for (int nf = 0; nf < 4; ++nf) {
        float pv = __expf(sf[nf][j] - mnew);
        sf[nf][j] = pv;
        psum += pv;
      }
#pragma unroll
      for (int d = 1; d < 16; d <<= 1) psum += __shfl_xor(psum, d, 32);
      float corr = __expf(mrow[j] - mnew);
      lrow[j] = lrow[j] * corr + psum;
#pragma unroll
      for (int nf = 0; nf < 4; ++nf) o[nf][j] *= corr;
      mrow[j] = mnew;
    }

    // C-layout -> A-layout relayout of P through per-wave LDS tile
#pragma unroll
    for (int nf = 0; nf < 4; ++nf)
#pragma unroll
      for (int j = 0; j < 8; ++j)
        Pw[wid][j + 8 * lh][nf * 16 + coll] = f2bf(sf[nf][j]);

    // O += P @ V   (B-frag column n = row n of transposed V tile)
#pragma unroll
    for (int ks = 0; ks < 2; ++ks) {
      v16bf pa = load_frag(&Pw[wid][0][ks * 32], 72);
#pragma unroll
      for (int nf = 0; nf < 4; ++nf)
        o[nf] = wmma_bf16(pa, load_frag(&Vt[nf * 16][ks * 32], 72), o[nf]);
    }
    __syncthreads();
  }

  __bf16* obase = O + ((size_t)(b * SS) + (size_t)qb * 128 + wid * 16) * DD + headoff;
#pragma unroll
  for (int nf = 0; nf < 4; ++nf)
#pragma unroll
    for (int j = 0; j < 8; ++j)
      obase[(size_t)(j + 8 * lh) * DD + nf * 16 + coll] = f2bf(o[nf][j] / lrow[j]);
}

// ---------------------------------------------------------------------------
extern "C" void kernel_launch(void* const* d_in, const int* in_sizes, int n_in,
                              void* d_out, int out_size, void* d_ws, size_t ws_size,
                              hipStream_t stream) {
  (void)in_sizes; (void)n_in; (void)out_size; (void)ws_size;
  const float* x_in  = (const float*)d_in[0];
  const float* Wq    = (const float*)d_in[1];
  const float* Wk    = (const float*)d_in[2];
  const float* Wv    = (const float*)d_in[3];
  const float* Wo    = (const float*)d_in[4];
  const float* W1    = (const float*)d_in[5];
  const float* b1    = (const float*)d_in[6];
  const float* W2    = (const float*)d_in[7];
  const float* b2    = (const float*)d_in[8];
  const float* alpha = (const float*)d_in[9];
  const float* beta  = (const float*)d_in[10];

  float* x = (float*)d_out;                 // fp32 residual stream in d_out
  const size_t BSD = (size_t)BB * SS * DD;  // 6291456
  const size_t WDD = (size_t)LL * DD * DD;  // per qkv/o weight tensor
  const size_t WFD = (size_t)LL * FF * DD;  // per ffn weight tensor

  __bf16* h   = (__bf16*)d_ws;
  __bf16* qb  = h  + BSD;
  __bf16* kb  = qb + BSD;
  __bf16* vb  = kb + BSD;
  __bf16* ao  = vb + BSD;
  __bf16* ffn = qb;                          // B*S*F == 4*BSD: aliases q..ao
  __bf16* wq  = ao + BSD;
  __bf16* wk  = wq + WDD;
  __bf16* wv  = wk + WDD;
  __bf16* wo  = wv + WDD;
  __bf16* w1  = wo + WDD;
  __bf16* w2  = w1 + WFD;

  hipMemcpyAsync(x, x_in, BSD * sizeof(float), hipMemcpyDeviceToDevice, stream);

  // One-time weight conversion to bf16 (halves HBM weight traffic)
  {
    int n4a = (int)(WDD / 4), n4b = (int)(WFD / 4);
    dim3 ga((n4a + 255) / 256), gb((n4b + 255) / 256), blk(256);
    f32_to_bf16<<<ga, blk, 0, stream>>>(Wq, wq, n4a);
    f32_to_bf16<<<ga, blk, 0, stream>>>(Wk, wk, n4a);
    f32_to_bf16<<<ga, blk, 0, stream>>>(Wv, wv, n4a);
    f32_to_bf16<<<ga, blk, 0, stream>>>(Wo, wo, n4a);
    f32_to_bf16<<<gb, blk, 0, stream>>>(W1, w1, n4b);
    f32_to_bf16<<<gb, blk, 0, stream>>>(W2, w2, n4b);
  }

  const int M = BB * SS;                    // 8192
  dim3 blk(256);
  dim3 lnb(32, 8);
  dim3 lng(DD / 32, BB);

  for (int i = 0; i < LL; ++i) {
    const float* al = alpha + (size_t)i * DD;
    const float* be = beta  + (size_t)i * DD;
    size_t wOff = (size_t)i * DD * DD;
    size_t fOff = (size_t)i * FF * DD;

    layernorm_seq<<<lng, lnb, 0, stream>>>(x, al, be, h);
    gemm_nt_bf16<<<dim3(DD / 128, M / 256), blk, 0, stream>>>(
        h, wq + wOff, qb, nullptr, nullptr, M, DD, DD, /*bf16 out*/2);
    gemm_nt_bf16<<<dim3(DD / 128, M / 256), blk, 0, stream>>>(
        h, wk + wOff, kb, nullptr, nullptr, M, DD, DD, 2);
    gemm_nt_bf16<<<dim3(DD / 128, M / 256), blk, 0, stream>>>(
        h, wv + wOff, vb, nullptr, nullptr, M, DD, DD, 2);
    flash_attn<<<dim3(SS / 128, BB * HH), blk, 0, stream>>>(qb, kb, vb, ao);
    gemm_nt_bf16<<<dim3(DD / 128, M / 256), blk, 0, stream>>>(
        ao, wo + wOff, x, nullptr, x, M, DD, DD, /*fp32+residual*/0);

    layernorm_seq<<<lng, lnb, 0, stream>>>(x, al, be, h);
    gemm_nt_bf16<<<dim3(FF / 128, M / 256), blk, 0, stream>>>(
        h, w1 + fOff, ffn, b1 + (size_t)i * FF, nullptr, M, FF, DD,
        /*gelu + bf16 out*/3);
    gemm_nt_bf16<<<dim3(DD / 128, M / 256), blk, 0, stream>>>(
        ffn, w2 + fOff, x, b2 + (size_t)i * DD, x, M, DD, FF, 0);
  }
}
